// CTBPRNStep_2817498546853
// MI455X (gfx1250) — compile-verified, matrix operand order checked by reading, not run on hardware
//
#include <hip/hip_runtime.h>
#include <math.h>

// ---------------------------------------------------------------------------
// CDNA5 (gfx1250, wave32) implementation of CTBPRNStep.
// Heavy GEMMs: v_wmma_f32_16x16x32_bf16 (fp32 accumulate).
// A tiles staged in LDS once per block (ds_load_b128 in inner loop);
// B tiles pre-packed to WMMA lane layout (pure global_load_b128, L2-resident).
// ---------------------------------------------------------------------------

typedef __attribute__((ext_vector_type(16))) __bf16 v16bf;
typedef __attribute__((ext_vector_type(8)))  __bf16 v8bf;
typedef __attribute__((ext_vector_type(8)))  float  v8f;

#define KK   3
#define BB   8
#define TT   1024
#define DD   1024
#define RRK  32
#define ROWS (BB*TT)          // 8192 rows per area
#define ASTR 1032             // LDS A row stride (8-elem pad -> bank-conflict-free)

// ---------------- scalar helpers ----------------
__device__ __forceinline__ unsigned short f2u16(float f) {
  unsigned u = __builtin_bit_cast(unsigned, f);
  u += 0x7fffu + ((u >> 16) & 1u);            // round-to-nearest-even
  return (unsigned short)(u >> 16);
}
__device__ __forceinline__ __bf16 f2bf(float f) {
  unsigned short s = f2u16(f);
  return __builtin_bit_cast(__bf16, s);
}
__device__ __forceinline__ float u16f(unsigned short s) {
  unsigned u = ((unsigned)s) << 16;
  return __builtin_bit_cast(float, u);
}
__device__ __forceinline__ float gelu_f(float x) {
  return 0.5f * x * (1.f + erff(x * 0.70710678118654752f));
}
__device__ __forceinline__ float sigm_f(float x) { return 1.f / (1.f + expf(-x)); }

__device__ __forceinline__ v8f wmma_bf16(v16bf a, v16bf b, v8f c) {
  // v_wmma_f32_16x16x32_bf16  (8-arg form, probe-confirmed)
  return __builtin_amdgcn_wmma_f32_16x16x32_bf16(false, a, false, b, (short)0, c,
                                                 false, false);
}

// ---------------- WMMA operand loaders ----------------
// A tile: 16 rows x 32 K.  lane: row m = lane&15, group = lane>>4.
// element e (0..15): k_local = (e<8 ? group*8+e : 16+group*8+(e-8))
__device__ __forceinline__ v16bf load_a_f32(const float* __restrict__ base, int ld,
                                            int row0, int kb, int lane, float scale) {
  int group = lane >> 4, m = lane & 15;
  const float* p = base + (long)(row0 + m) * ld + kb + group * 8;
  float4 f0 = *(const float4*)(p);
  float4 f1 = *(const float4*)(p + 4);
  float4 f2 = *(const float4*)(p + 16);
  float4 f3 = *(const float4*)(p + 20);
  v16bf a;
  a[0]  = f2bf(f0.x*scale); a[1]  = f2bf(f0.y*scale); a[2]  = f2bf(f0.z*scale); a[3]  = f2bf(f0.w*scale);
  a[4]  = f2bf(f1.x*scale); a[5]  = f2bf(f1.y*scale); a[6]  = f2bf(f1.z*scale); a[7]  = f2bf(f1.w*scale);
  a[8]  = f2bf(f2.x*scale); a[9]  = f2bf(f2.y*scale); a[10] = f2bf(f2.z*scale); a[11] = f2bf(f2.w*scale);
  a[12] = f2bf(f3.x*scale); a[13] = f2bf(f3.y*scale); a[14] = f2bf(f3.z*scale); a[15] = f2bf(f3.w*scale);
  return a;
}
// Read an A fragment from the LDS-staged tile (ds_load_b128 x2).
__device__ __forceinline__ v16bf lds_a(const unsigned short* sm, int kb, int lane) {
  int group = lane >> 4, m = lane & 15;
  const unsigned short* p = sm + m * ASTR + kb + group * 8;
  v8bf lo = *(const v8bf*)(p);
  v8bf hi = *(const v8bf*)(p + 16);
  return __builtin_shufflevector(lo, hi, 0,1,2,3,4,5,6,7,8,9,10,11,12,13,14,15);
}
// Cooperative copy of a 16 x din bf16 tile into LDS (padded rows).
__device__ __forceinline__ void stage_a(const unsigned short* __restrict__ g, int ld,
                                        int row0, int din, unsigned short* sm) {
  int nch = din >> 3;                       // 8-elem (16B) chunks per row
  int tot = nch << 4;                       // 16 rows
  for (int idx = threadIdx.x; idx < tot; idx += blockDim.x) {
    int r = idx / nch, c = idx - r * nch;
    *(uint4*)(sm + r * ASTR + c * 8) =
        *(const uint4*)(g + (long)(row0 + r) * ld + c * 8);
  }
}
// B pack: tile (nt,kt) stored as 32 lanes x 16 bf16 contiguous (1 KB / tile).
__device__ __forceinline__ v16bf load_b_pack(const unsigned short* __restrict__ pack,
                                             int numKt, int nt, int kt, int lane) {
  return *(const v16bf*)(pack + (((long)(nt * numKt + kt)) << 9) + (lane << 4));
}

// ---------------- X fp32 -> bf16 bulk convert (one pass) ----------------
__global__ void k_cvt(const float* __restrict__ X, unsigned short* __restrict__ Xb, long n8) {
  long i = (long)blockIdx.x * blockDim.x + threadIdx.x;    // one thread per 8 elems
  if (i >= n8) return;
  const float4* p = (const float4*)X + i * 2;
  float4 a = p[0], b = p[1];
  uint4 o;
  o.x = (unsigned)f2u16(a.x) | ((unsigned)f2u16(a.y) << 16);
  o.y = (unsigned)f2u16(a.z) | ((unsigned)f2u16(a.w) << 16);
  o.z = (unsigned)f2u16(b.x) | ((unsigned)f2u16(b.y) << 16);
  o.w = (unsigned)f2u16(b.z) | ((unsigned)f2u16(b.w) << 16);
  ((uint4*)Xb)[i] = o;
}

// ---------------- weight packing (fp32 [din,dout] -> WMMA-B bf16) ----------------
__global__ void k_pack_w(const float* __restrict__ src, unsigned short* __restrict__ dst,
                         int din, int dout, long srcK, long dstK) {
  int k   = blockIdx.y;
  int idx = blockIdx.x * blockDim.x + threadIdx.x;
  int numKt = din >> 5;
  int total = numKt * (dout >> 4) * 512;
  if (idx >= total) return;
  int e    = idx & 15;
  int lane = (idx >> 4) & 31;
  int tile = idx >> 9;
  int nt = tile / numKt, kt = tile - nt * numKt;
  int group = lane >> 4, n = lane & 15;
  int klocal = (e < 8) ? (group * 8 + e) : (16 + group * 8 + (e - 8));
  float v = src[(long)k * srcK + (long)(kt * 32 + klocal) * dout + nt * 16 + n];
  dst[(long)k * dstK + idx] = f2u16(v);
}
// packed B[k][n] = src[n*ldsrc + k]  (used for U^T: msg = low @ U^T)
__global__ void k_pack_wt(const float* __restrict__ src, unsigned short* __restrict__ dst,
                          int din, int dout, int ldsrc, long srcK, long dstK) {
  int k   = blockIdx.y;
  int idx = blockIdx.x * blockDim.x + threadIdx.x;
  int numKt = din >> 5;
  int total = numKt * (dout >> 4) * 512;
  if (idx >= total) return;
  int e    = idx & 15;
  int lane = (idx >> 4) & 31;
  int tile = idx >> 9;
  int nt = tile / numKt, kt = tile - nt * numKt;
  int group = lane >> 4, n = lane & 15;
  int klocal = (e < 8) ? (group * 8 + e) : (16 + group * 8 + (e - 8));
  float v = src[(long)k * srcK + (long)(nt * 16 + n) * ldsrc + kt * 32 + klocal];
  dst[(long)k * dstK + idx] = f2u16(v);
}

// ---------------- generic WMMA GEMM:  Out = act(A @ Bpack), A bf16 via LDS ----------------
// NOTE: launch configs cover dout exactly (no partial waves) -> barriers are safe.
template<int NT, bool ACT_GELU, bool OBF16>
__global__ void k_gemm(const unsigned short* __restrict__ A, long aK,
                       const unsigned short* __restrict__ Bp, long bK,
                       void* __restrict__ Out, long oK, int din, int dout) {
  __shared__ unsigned short smA[16 * ASTR];
  int k    = blockIdx.z;
  int lane = threadIdx.x & 31;
  int wave = threadIdx.x >> 5;
  int waves = blockDim.x >> 5;
  int row0 = blockIdx.x << 4;
  int nt0  = (blockIdx.y * waves + wave) * NT;
  int numKt = din >> 5;
  const unsigned short* bp = Bp + (long)k * bK;
  stage_a(A + (long)k * aK, din, row0, din, smA);
  __syncthreads();
  v8f zero = {0.f,0.f,0.f,0.f,0.f,0.f,0.f,0.f};
  v8f acc[NT];
#pragma unroll
  for (int j = 0; j < NT; ++j) acc[j] = zero;
  for (int kt = 0; kt < numKt; ++kt) {
    if (kt + 1 < numKt)   // gfx1250 global_prefetch of next B tile (L2-resident)
      __builtin_prefetch(bp + (((long)(nt0 * numKt + kt + 1)) << 9) + (lane << 4), 0, 1);
    v16bf a = lds_a(smA, kt * 32, lane);
#pragma unroll
    for (int j = 0; j < NT; ++j) {
      v16bf bm = load_b_pack(bp, numKt, nt0 + j, kt, lane);
      acc[j] = wmma_bf16(a, bm, acc[j]);
    }
  }
  int group = lane >> 4, n = lane & 15;
#pragma unroll
  for (int j = 0; j < NT; ++j)
#pragma unroll
    for (int i = 0; i < 8; ++i) {
      float v = acc[j][i];
      if (ACT_GELU) v = gelu_f(v);
      long off = (long)(row0 + group * 8 + i) * dout + (nt0 + j) * 16 + n;
      if (OBF16) ((unsigned short*)Out)[(long)k * oK + off] = f2u16(v);
      else       ((float*)Out)[(long)k * oK + off] = v;
    }
}

// ---------------- routed = sum_k alpha[b,k] * (low_k @ U_k^T)  (bf16 out) ----------------
__global__ void k_routed(const float* __restrict__ low, const unsigned short* __restrict__ pU,
                         const float* __restrict__ alpha, unsigned short* __restrict__ routed) {
  constexpr int NT = 2;
  int lane = threadIdx.x & 31;
  int wave = threadIdx.x >> 5;
  int waves = blockDim.x >> 5;
  int row0 = blockIdx.x << 4;
  int nt0  = (blockIdx.y * waves + wave) * NT;
  int b    = row0 >> 10;                       // T = 1024 rows per batch
  v8f zero = {0.f,0.f,0.f,0.f,0.f,0.f,0.f,0.f};
  v8f acc[NT]; acc[0] = zero; acc[1] = zero;
  for (int k = 0; k < KK; ++k) {
    float al = alpha[b * KK + k];
    v16bf a = load_a_f32(low + (long)k * ROWS * RRK, RRK, row0, 0, lane, al);
#pragma unroll
    for (int j = 0; j < NT; ++j) {
      v16bf bm = load_b_pack(pU + (long)k * (64L * 512), 1, nt0 + j, 0, lane);
      acc[j] = wmma_bf16(a, bm, acc[j]);
    }
  }
  int group = lane >> 4, n = lane & 15;
#pragma unroll
  for (int j = 0; j < NT; ++j)
#pragma unroll
    for (int i = 0; i < 8; ++i)
      routed[(long)(row0 + group * 8 + i) * DD + (nt0 + j) * 16 + n] = f2u16(acc[j][i]);
}

// ---------------- fused column epilogue (writes pre-LN result into y) ----------------
__global__ void k_final(const float* __restrict__ X, const unsigned short* __restrict__ Xbf,
                        const unsigned short* __restrict__ routed,
                        const unsigned short* __restrict__ bcomp, const unsigned short* __restrict__ ah,
                        const unsigned short* __restrict__ pWg, const unsigned short* __restrict__ pWm1,
                        const unsigned short* __restrict__ pWm2, const float* __restrict__ bg,
                        float* __restrict__ y) {
  __shared__ unsigned short smA[16 * ASTR];
  constexpr int NT = 2;
  int k    = blockIdx.z;
  int lane = threadIdx.x & 31;
  int wave = threadIdx.x >> 5;
  int waves = blockDim.x >> 5;
  int row0 = blockIdx.x << 4;
  int nt0  = (blockIdx.y * waves + wave) * NT;
  const long KRD = (long)ROWS * DD;
  const float* Xk           = X      + (long)k * KRD;
  const unsigned short* Xbk = Xbf    + (long)k * KRD;
  const unsigned short* bck = bcomp  + (long)k * KRD;
  const unsigned short* ahk = ah     + (long)k * KRD;
  const unsigned short* wg  = pWg    + (long)k * (64L * 64 * 512);
  const unsigned short* wm1 = pWm1   + (long)k * (32L * 64 * 512);
  const unsigned short* wm2 = pWm2   + (long)k * (32L * 64 * 512);
  v8f zero = {0.f,0.f,0.f,0.f,0.f,0.f,0.f,0.f};
  v8f ag[NT], am1[NT], am2[NT];
#pragma unroll
  for (int j = 0; j < NT; ++j) { ag[j] = zero; am1[j] = zero; am2[j] = zero; }
  // gate: X @ Wg_top  (Wg rows 0..1023)
  stage_a(Xbk, DD, row0, DD, smA);
  __syncthreads();
  for (int kt = 0; kt < 32; ++kt) {
    __builtin_prefetch(wg + (((long)(nt0 * 64 + kt + 1)) << 9) + (lane << 4), 0, 1);
    v16bf a = lds_a(smA, kt * 32, lane);
#pragma unroll
    for (int j = 0; j < NT; ++j)
      ag[j] = wmma_bf16(a, load_b_pack(wg, 64, nt0 + j, kt, lane), ag[j]);
  }
  __syncthreads();
  // gate: routed @ Wg_bot  (Wg rows 1024..2047)
  stage_a(routed, DD, row0, DD, smA);
  __syncthreads();
  for (int kt = 0; kt < 32; ++kt) {
    v16bf a = lds_a(smA, kt * 32, lane);
#pragma unroll
    for (int j = 0; j < NT; ++j)
      ag[j] = wmma_bf16(a, load_b_pack(wg, 64, nt0 + j, 32 + kt, lane), ag[j]);
  }
  __syncthreads();
  // m1: bcomp @ Wm1
  stage_a(bck, DD, row0, DD, smA);
  __syncthreads();
  for (int kt = 0; kt < 32; ++kt) {
    v16bf a = lds_a(smA, kt * 32, lane);
#pragma unroll
    for (int j = 0; j < NT; ++j)
      am1[j] = wmma_bf16(a, load_b_pack(wm1, 32, nt0 + j, kt, lane), am1[j]);
  }
  __syncthreads();
  // m2: ah @ Wm2
  stage_a(ahk, DD, row0, DD, smA);
  __syncthreads();
  for (int kt = 0; kt < 32; ++kt) {
    v16bf a = lds_a(smA, kt * 32, lane);
#pragma unroll
    for (int j = 0; j < NT; ++j)
      am2[j] = wmma_bf16(a, load_b_pack(wm2, 32, nt0 + j, kt, lane), am2[j]);
  }
  int group = lane >> 4, n = lane & 15;
#pragma unroll
  for (int j = 0; j < NT; ++j)
#pragma unroll
    for (int i = 0; i < 8; ++i) {
      int  rr = row0 + group * 8 + i;
      int  cc = (nt0 + j) * 16 + n;
      float x  = Xk[(long)rr * DD + cc];
      float bc = u16f(bck[(long)rr * DD + cc]);
      float gg = sigm_f(ag[j][i] + bg[k * DD + cc]);
      float m  = gelu_f(am1[j][i]) * gelu_f(am2[j][i]);
      y[(long)k * KRD + (long)rr * DD + cc] = x + (1.f - gg) * bc + gg * m;
    }
}

// ---------------- LayerNorm (in place, per row of 1024, per-k gamma/beta) ----------------
__global__ void k_ln_final(float* __restrict__ y, const float* __restrict__ g,
                           const float* __restrict__ b) {
  __shared__ float red[256];
  int row = blockIdx.x;                 // 0 .. K*ROWS-1
  int k   = row >> 13;                  // / 8192
  float* p = y + (long)row * DD;
  float s = 0.f;
  for (int i = threadIdx.x; i < DD; i += blockDim.x) s += p[i];
  red[threadIdx.x] = s; __syncthreads();
  for (int st = 128; st > 0; st >>= 1) { if (threadIdx.x < st) red[threadIdx.x] += red[threadIdx.x + st]; __syncthreads(); }
  float mu = red[0] * (1.f / DD);
  __syncthreads();
  float v = 0.f;
  for (int i = threadIdx.x; i < DD; i += blockDim.x) { float d0 = p[i] - mu; v += d0 * d0; }
  red[threadIdx.x] = v; __syncthreads();
  for (int st = 128; st > 0; st >>= 1) { if (threadIdx.x < st) red[threadIdx.x] += red[threadIdx.x + st]; __syncthreads(); }
  float rs = rsqrtf(red[0] * (1.f / DD) + 1e-5f);
  __syncthreads();
  for (int i = threadIdx.x; i < DD; i += blockDim.x)
    p[i] = (p[i] - mu) * rs * g[k * DD + i] + b[k * DD + i];
}

// ---------------- BGController (small) ----------------
__global__ void k_mean(const float* __restrict__ X, float* __restrict__ am) {
  int idx = blockIdx.x * blockDim.x + threadIdx.x;
  if (idx >= KK * BB * DD) return;
  int e = idx % DD, b = (idx / DD) % BB, k = idx / (BB * DD);
  const float* p = X + ((long)(k * BB + b) * TT) * DD + e;
  float s = 0.f;
  for (int t = 0; t < TT; ++t) s += p[(long)t * DD];
  am[((long)b * KK + k) * DD + e] = s * (1.f / TT);
}
__global__ void k_proj(const float* __restrict__ am, const float* __restrict__ W,
                       const float* __restrict__ bias, float* __restrict__ pm) {
  int idx = blockIdx.x * blockDim.x + threadIdx.x;
  if (idx >= BB * KK * DD) return;
  int row = idx / DD, e = idx % DD;
  const float* a = am + (long)row * DD;
  float s = bias[e];
  for (int i = 0; i < DD; ++i) s += a[i] * W[(long)i * DD + e];
  pm[idx] = gelu_f(s);
}
__global__ void k_headln(const float* __restrict__ pm, const float* __restrict__ g,
                         const float* __restrict__ b, float* __restrict__ h) {
  __shared__ float red[256];
  const int L = KK * DD;
  int bi = blockIdx.x;
  const float* p = pm + (long)bi * L;
  float s = 0.f;
  for (int i = threadIdx.x; i < L; i += blockDim.x) s += p[i];
  red[threadIdx.x] = s; __syncthreads();
  for (int st = 128; st > 0; st >>= 1) { if (threadIdx.x < st) red[threadIdx.x] += red[threadIdx.x + st]; __syncthreads(); }
  float mu = red[0] / (float)L;
  __syncthreads();
  float v = 0.f;
  for (int i = threadIdx.x; i < L; i += blockDim.x) { float d0 = p[i] - mu; v += d0 * d0; }
  red[threadIdx.x] = v; __syncthreads();
  for (int st = 128; st > 0; st >>= 1) { if (threadIdx.x < st) red[threadIdx.x] += red[threadIdx.x + st]; __syncthreads(); }
  float rs = rsqrtf(red[0] / (float)L + 1e-5f);
  __syncthreads();
  for (int i = threadIdx.x; i < L; i += blockDim.x)
    h[(long)bi * L + i] = (p[i] - mu) * rs * g[i] + b[i];
}
__global__ void k_head1(const float* __restrict__ h, const float* __restrict__ W1,
                        const float* __restrict__ b1, float* __restrict__ h1) {
  int idx = blockIdx.x * blockDim.x + threadIdx.x;
  const int Din = KK * DD, Dout = 2 * DD;
  if (idx >= BB * Dout) return;
  int b = idx / Dout, j = idx % Dout;
  const float* p = h + (long)b * Din;
  float s = b1[j];
  for (int i = 0; i < Din; ++i) s += p[i] * W1[(long)i * Dout + j];
  h1[idx] = gelu_f(s);
}
__global__ void k_head2(const float* __restrict__ h1, const float* __restrict__ W2,
                        const float* __restrict__ b2, float* __restrict__ logits) {
  int t = threadIdx.x;
  if (t >= BB * KK) return;
  int b = t / KK, kk = t % KK;
  const int Din = 2 * DD;
  const float* p = h1 + (long)b * Din;
  float s = b2[kk];
  for (int j = 0; j < Din; ++j) s += p[j] * W2[(long)j * KK + kk];
  logits[t] = s;
}
__global__ void k_alpha(const float* __restrict__ logits, const float* __restrict__ gumbel,
                        float* __restrict__ alpha) {
  int t = threadIdx.x;
  if (t >= BB * KK) return;
  int b = t / KK, kk = t % KK;
  float z[KK]; float mx = -1e30f;
  for (int i = 0; i < KK; ++i) {
    z[i] = (logits[b * KK + i] + gumbel[b * KK + i]) * 1.25f;   // / TAU(0.8)
    mx = fmaxf(mx, z[i]);
  }
  float s = 0.f;
  for (int i = 0; i < KK; ++i) s += expf(z[i] - mx);
  alpha[t] = expf(z[kk] - mx) / s;
}

// ---------------------------------------------------------------------------
extern "C" void kernel_launch(void* const* d_in, const int* in_sizes, int n_in,
                              void* d_out, int out_size, void* d_ws, size_t ws_size,
                              hipStream_t stream) {
  const float* X      = (const float*)d_in[0];
  const float* gumbel = (const float*)d_in[1];
  const float* proj_W = (const float*)d_in[2];
  const float* proj_b = (const float*)d_in[3];
  const float* hln_g  = (const float*)d_in[4];
  const float* hln_b  = (const float*)d_in[5];
  const float* W1     = (const float*)d_in[6];
  const float* b1     = (const float*)d_in[7];
  const float* W2     = (const float*)d_in[8];
  const float* b2     = (const float*)d_in[9];
  const float* U      = (const float*)d_in[10];
  const float* V      = (const float*)d_in[11];
  const float* Wb     = (const float*)d_in[12];
  const float* Wa     = (const float*)d_in[13];
  const float* Wg     = (const float*)d_in[14];
  const float* bg     = (const float*)d_in[15];
  const float* Wm1    = (const float*)d_in[16];
  const float* Wm2    = (const float*)d_in[17];
  const float* cln_g  = (const float*)d_in[18];
  const float* cln_b  = (const float*)d_in[19];

  float* y      = (float*)d_out;                       // [K,B,T,d]
  float* logits = y + (long)KK * ROWS * DD;            // [B,K] tail

  // ---- workspace carve-up ----
  char* w = (char*)d_ws;
  auto take = [&](size_t bytes) { char* r = w; w += (bytes + 255) & ~(size_t)255; return r; };
  const long SZ_PACK_DD = 3L * 32 * 64 * 512;          // ushorts per d x d family
  const long SZ_PACK_WG = 3L * 64 * 64 * 512;
  const long SZ_PACK_V  = 3L * 32 * 2  * 512;
  const long SZ_PACK_U  = 3L * 1  * 64 * 512;
  unsigned short* pWb  = (unsigned short*)take(SZ_PACK_DD * 2);
  unsigned short* pWa  = (unsigned short*)take(SZ_PACK_DD * 2);
  unsigned short* pWm1 = (unsigned short*)take(SZ_PACK_DD * 2);
  unsigned short* pWm2 = (unsigned short*)take(SZ_PACK_DD * 2);
  unsigned short* pWg  = (unsigned short*)take(SZ_PACK_WG * 2);
  unsigned short* pV   = (unsigned short*)take(SZ_PACK_V * 2);
  unsigned short* pU   = (unsigned short*)take(SZ_PACK_U * 2);
  unsigned short* Xbf    = (unsigned short*)take(3L * ROWS * DD * 2);
  float*          low    = (float*)take(3L * ROWS * RRK * 4);
  unsigned short* routed = (unsigned short*)take((long)ROWS * DD * 2);
  unsigned short* bcomp  = (unsigned short*)take(3L * ROWS * DD * 2);
  unsigned short* ah     = (unsigned short*)take(3L * ROWS * DD * 2);
  float* am    = (float*)take(BB * KK * DD * 4);
  float* pm    = (float*)take(BB * KK * DD * 4);
  float* h     = (float*)take(BB * KK * DD * 4);
  float* h1    = (float*)take(BB * 2 * DD * 4);
  float* alpha = (float*)take(BB * KK * 4);
  (void)in_sizes; (void)n_in; (void)out_size; (void)ws_size;

  // ---- 1. pack weights into WMMA-B bf16 layouts; bulk-convert X ----
  k_pack_w <<<dim3(4096, KK), 256, 0, stream>>>(Wb,  pWb,  DD,   DD, (long)DD * DD,       32L * 64 * 512);
  k_pack_w <<<dim3(4096, KK), 256, 0, stream>>>(Wa,  pWa,  DD,   DD, (long)DD * DD,       32L * 64 * 512);
  k_pack_w <<<dim3(4096, KK), 256, 0, stream>>>(Wm1, pWm1, DD,   DD, (long)DD * DD,       32L * 64 * 512);
  k_pack_w <<<dim3(4096, KK), 256, 0, stream>>>(Wm2, pWm2, DD,   DD, (long)DD * DD,       32L * 64 * 512);
  k_pack_w <<<dim3(8192, KK), 256, 0, stream>>>(Wg,  pWg,  2*DD, DD, (long)2 * DD * DD,   64L * 64 * 512);
  k_pack_w <<<dim3(128,  KK), 256, 0, stream>>>(V,   pV,   DD,  RRK, (long)DD * RRK,      32L * 2 * 512);
  k_pack_wt<<<dim3(128,  KK), 256, 0, stream>>>(U,   pU,   RRK,  DD, RRK, (long)DD * RRK, 64L * 512);
  {
    long n8 = (3L * ROWS * DD) / 8;                    // 3,145,728
    k_cvt<<<(unsigned)((n8 + 255) / 256), 256, 0, stream>>>(X, Xbf, n8);
  }

  // ---- 2. controller (tiny) ----
  k_mean  <<<96, 256, 0, stream>>>(X, am);
  k_proj  <<<96, 256, 0, stream>>>(am, proj_W, proj_b, pm);
  k_headln<<<BB, 256, 0, stream>>>(pm, hln_g, hln_b, h);
  k_head1 <<<64, 256, 0, stream>>>(h, W1, b1, h1);
  k_head2 <<<1, 32, 0, stream>>>(h1, W2, b2, logits);
  k_alpha <<<1, 32, 0, stream>>>(logits, gumbel, alpha);

  // ---- 3. router ----
  k_gemm<2, false, false><<<dim3(512, 1, KK), 32, 0, stream>>>(
      Xbf, (long)ROWS * DD, pV, 32L * 2 * 512, low, (long)ROWS * RRK, DD, RRK);
  k_routed<<<dim3(512, 8), 128, 0, stream>>>(low, pU, alpha, routed);

  // ---- 4. column branches ----
  k_gemm<4, true, true><<<dim3(512, 4, KK), 128, 0, stream>>>(
      Xbf, (long)ROWS * DD, pWb, 32L * 64 * 512, bcomp, (long)ROWS * DD, DD, DD);
  k_gemm<4, true, true><<<dim3(512, 4, KK), 128, 0, stream>>>(
      routed, 0L, pWa, 32L * 64 * 512, ah, (long)ROWS * DD, DD, DD);

  // ---- 5. fused gate/modulation epilogue + LayerNorm ----
  k_final<<<dim3(512, 8, KK), 128, 0, stream>>>(X, Xbf, routed, bcomp, ah, pWg, pWm1, pWm2, bg, y);
  k_ln_final<<<KK * ROWS, 256, 0, stream>>>(y, cln_g, cln_b);
}
